// CharRNN_86792699118081
// MI455X (gfx1250) — matrix operand order repaired and checked by reading, not compiled, not run on hardware
//
#include <hip/hip_runtime.h>

// ---------------------------------------------------------------------------
// CharRNN fused persistent kernel for MI455X (gfx1250, wave32, WMMA).
//   B=64, L=512, E=128, H=1024, V=256
// One 1024-thread workgroup keeps h[64][1024] resident in LDS (320KB/WGP),
// runs the 512 sequential steps fully fused (embed -> x-proj -> h@W_hh ->
// tanh -> logits), using v_wmma_f32_16x16x32_bf16 with f32 accumulation.
// Weights + embedding table are pre-converted once to bf16; weights are
// pre-swizzled into WMMA B-fragments (contiguous k: 16*half + elem) so the hot loop
// does only coalesced loads. Embedding rows are gathered into LDS each step
// with GLOBAL_LOAD_ASYNC_TO_LDS_B128 (ASYNCcnt path).
// ---------------------------------------------------------------------------

#define B_ 64
#define L_ 512
#define E_ 128
#define H_ 1024
#define V_ 256

typedef __bf16 bf16_t;
typedef __attribute__((ext_vector_type(16))) __bf16 v16bf;   // 32 bytes
typedef __attribute__((ext_vector_type(8)))  __bf16 v8bf;    // 16 bytes
typedef __attribute__((ext_vector_type(8)))  float  v8f;
typedef __attribute__((__vector_size__(4 * sizeof(int)))) int v4i; // 16 bytes

#define AS1 __attribute__((address_space(1)))
#define AS3 __attribute__((address_space(3)))

#define HS 1032   // padded LDS row stride (halves) for h   (2064B, 16B-mult)
#define ES 136    // padded LDS row stride (halves) for emb (272B, 16B-mult)

// Swizzled offset (in halves) of element (k,n) of a KxN weight matrix, laid
// out so a wave's B-fragment for one 16x16xK32 WMMA is 32 contiguous bytes
// per lane:  [k-tile][n-tile][lane][elem]
//   lane  = ((k>>4)&1)*16 + (n&15)    (contiguous k order, wave32)
//   elem  = k & 15                    (VGPR j holds K=2j,2j+1)
__device__ __forceinline__ size_t swz_off(int k, int n, int N) {
  return (size_t)(k >> 5) * (size_t)(32 * N)   // K-tile block
       + (size_t)(n >> 4) * 512u               // N-tile block (32 lanes * 16)
       + (size_t)((k >> 4) & 1) * 256u         // upper half-wave lanes
       + (size_t)(n & 15) * 16u                // lane within half
       + (size_t)(k & 15);                     // element
}

__global__ void swizzle_weights(const float* __restrict__ W,
                                bf16_t* __restrict__ Wz, int K, int N) {
  int total = K * N;
  for (int i = blockIdx.x * blockDim.x + threadIdx.x; i < total;
       i += gridDim.x * blockDim.x) {
    int k = i / N, n = i - k * N;
    Wz[swz_off(k, n, N)] = (bf16_t)W[i];
  }
}

// Plain row-major f32 -> bf16 conversion (embedding table, 64KB result).
__global__ void convert_bf16(const float* __restrict__ src,
                             bf16_t* __restrict__ dst, int total) {
  for (int i = blockIdx.x * blockDim.x + threadIdx.x; i < total;
       i += gridDim.x * blockDim.x)
    dst[i] = (bf16_t)src[i];
}

// A-fragment (16x32 bf16): lanes 0-15 row M=lane hold K=0..15,
// lanes 16-31 hold K=16..31. rowbase points at &buf[m*stride + k0 + kh8].
__device__ __forceinline__ v16bf load_a_frag(const bf16_t* rowbase) {
  v8bf lo = *(const v8bf*)(rowbase);        // K = k0+kh8 .. +7
  v8bf hi = *(const v8bf*)(rowbase + 8);    // K = k0+kh8+8 .. +15
  v16bf a;
#pragma unroll
  for (int i = 0; i < 8; ++i) { a[i] = lo[i]; a[8 + i] = hi[i]; }
  return a;
}

__device__ __forceinline__ v8f wmma_bf16(v16bf a, v16bf b, v8f c) {
  return __builtin_amdgcn_wmma_f32_16x16x32_bf16(
      /*neg_a=*/false, a, /*neg_b=*/false, b,
      /*c_mod=*/(short)0, c, /*reuse_a=*/false, /*reuse_b=*/false);
}

// Fast, saturating tanh: 1 - 2/(exp2(2x*log2e)+1).  exp overflow -> +1,
// underflow -> -1, so it is robust for any accumulator value.
__device__ __forceinline__ float fast_tanh(float x) {
  float e = __builtin_amdgcn_exp2f(x * 2.8853900817779268f); // exp(2x)
  return 1.0f - 2.0f * __builtin_amdgcn_rcpf(e + 1.0f);
}

__global__ void __launch_bounds__(1024)
rnn_fused(const int* __restrict__ x, const float* __restrict__ hidden0,
          const bf16_t* __restrict__ emb_bf,
          const bf16_t* __restrict__ Wxh_z, const bf16_t* __restrict__ Whh_z,
          const float* __restrict__ b_h,
          const bf16_t* __restrict__ Why_z, const float* __restrict__ b_y,
          float* __restrict__ logits, float* __restrict__ final_h) {
  extern __shared__ char smem[];
  bf16_t* hbuf = (bf16_t*)smem;                                      // [64][HS]
  bf16_t* ebuf = (bf16_t*)(smem + (size_t)B_ * HS * sizeof(bf16_t)); // [64][ES]

  const int tid  = threadIdx.x;
  const int lane = tid & 31;
  const int wave = tid >> 5;         // 32 waves
  const int lm   = lane & 15;        // col / row-within-tile
  const int kh8  = (lane >> 4) * 16; // A-frag K offset for upper half-wave
  const int hi8  = (lane >> 4) * 8;  // C row offset for upper half-wave

  // --- tile ownership -------------------------------------------------------
  // GEMM1/2 (output [64][1024]): wave owns N-tiles {2w, 2w+1} x all 4 M-tiles.
  const int nt0 = wave * 2;
  // GEMM3 (output [64][256]): wave owns M-tile w>>3, N-tiles {(w&7)*2, +1}.
  const int mt3 = wave >> 3;
  const int nt3 = (wave & 7) * 2;

  // Hoisted biases (constant over t).
  const float bh0 = b_h[(nt0 + 0) * 16 + lm];
  const float bh1 = b_h[(nt0 + 1) * 16 + lm];
  const float by0 = b_y[(nt3 + 0) * 16 + lm];
  const float by1 = b_y[(nt3 + 1) * 16 + lm];

  // Per-thread async-gather assignment: thread copies 16B (8 bf16) of one
  // embedding row per step.  1024 threads x 16B == the whole 64x128 tile.
  const int eb = tid >> 4;           // batch row 0..63
  const int ee = (tid & 15) * 8;     // element offset 0..120
  bf16_t* edst = ebuf + eb * ES + ee;

  // --- init h from hidden0 --------------------------------------------------
  for (int i = tid; i < B_ * H_; i += 1024) {
    int b = i >> 10, n = i & (H_ - 1);
    hbuf[b * HS + n] = (bf16_t)hidden0[i];
  }
  __syncthreads();

  for (int t = 0; t < L_; ++t) {
    // ---- async-gather embedding rows for x[:, t] into LDS (bf16) ----------
    {
      int tok = x[eb * L_ + t];
      const bf16_t* esrc = emb_bf + tok * E_ + ee;
      __builtin_amdgcn_global_load_async_to_lds_b128(
          (AS1 v4i*)(v4i*)esrc, (AS3 v4i*)(v4i*)edst, 0, 0);
      asm volatile("s_wait_asynccnt 0x0" ::: "memory");
    }
    __syncthreads();

    // ---- acc = h_prev @ W_hh + emb @ W_xh + b_h ---------------------------
    v8f acc[2][4];
#pragma unroll
    for (int j = 0; j < 2; ++j)
#pragma unroll
      for (int mt = 0; mt < 4; ++mt) acc[j][mt] = (v8f)0.0f;

#pragma unroll 4
    for (int kt = 0; kt < H_ / 32; ++kt) {              // K = 1024
      const bf16_t* wb = Whh_z + (size_t)kt * (32 * H_) + (size_t)lane * 16;
      v16bf b0 = *(const v16bf*)(wb + (size_t)(nt0 + 0) * 512);
      v16bf b1 = *(const v16bf*)(wb + (size_t)(nt0 + 1) * 512);
      // warm the next K-tile of this wave's weight stream in L2/L0
      __builtin_prefetch(wb + (size_t)(32 * H_) + (size_t)nt0 * 512, 0, 1);
#pragma unroll
      for (int mt = 0; mt < 4; ++mt) {
        v16bf a = load_a_frag(&hbuf[(mt * 16 + lm) * HS + kt * 32 + kh8]);
        acc[0][mt] = wmma_bf16(a, b0, acc[0][mt]);
        acc[1][mt] = wmma_bf16(a, b1, acc[1][mt]);
      }
    }
#pragma unroll
    for (int kt = 0; kt < E_ / 32; ++kt) {              // K = 128
      const bf16_t* wb = Wxh_z + (size_t)kt * (32 * H_) + (size_t)lane * 16;
      v16bf b0 = *(const v16bf*)(wb + (size_t)(nt0 + 0) * 512);
      v16bf b1 = *(const v16bf*)(wb + (size_t)(nt0 + 1) * 512);
#pragma unroll
      for (int mt = 0; mt < 4; ++mt) {
        v16bf a = load_a_frag(&ebuf[(mt * 16 + lm) * ES + kt * 32 + kh8]);
        acc[0][mt] = wmma_bf16(a, b0, acc[0][mt]);
        acc[1][mt] = wmma_bf16(a, b1, acc[1][mt]);
      }
    }
    __syncthreads();   // everyone is done READING h_prev

    // ---- h_t = tanh(acc + b_h); write in place (bf16) ---------------------
#pragma unroll
    for (int j = 0; j < 2; ++j) {
      const float bias = j ? bh1 : bh0;
      const int col = (nt0 + j) * 16 + lm;
#pragma unroll
      for (int mt = 0; mt < 4; ++mt) {
#pragma unroll
        for (int r = 0; r < 8; ++r) {
          float v = fast_tanh(acc[j][mt][r] + bias);
          int row = mt * 16 + hi8 + r;
          hbuf[row * HS + col] = (bf16_t)v;
          if (t == L_ - 1) final_h[row * H_ + col] = v;   // uniform branch
        }
      }
    }
    __syncthreads();   // h_t visible to all waves

    // ---- logits_t = h_t @ W_hy + b_y  -> d_out ----------------------------
    v8f c0 = (v8f)0.0f, c1 = (v8f)0.0f;
#pragma unroll 4
    for (int kt = 0; kt < H_ / 32; ++kt) {
      const bf16_t* wb = Why_z + (size_t)kt * (32 * V_) + (size_t)lane * 16;
      v16bf b0 = *(const v16bf*)(wb + (size_t)(nt3 + 0) * 512);
      v16bf b1 = *(const v16bf*)(wb + (size_t)(nt3 + 1) * 512);
      v16bf a  = load_a_frag(&hbuf[(mt3 * 16 + lm) * HS + kt * 32 + kh8]);
      c0 = wmma_bf16(a, b0, c0);
      c1 = wmma_bf16(a, b1, c1);
    }
#pragma unroll
    for (int r = 0; r < 8; ++r) {
      int row = mt3 * 16 + hi8 + r;                     // batch index b
      size_t base = ((size_t)row * L_ + t) * V_;
      logits[base + (nt3 + 0) * 16 + lm] = c0[r] + by0;
      logits[base + (nt3 + 1) * 16 + lm] = c1[r] + by1;
    }
    // No barrier needed: next-step async ebuf staging does not touch hbuf,
    // and the next __syncthreads orders it before any h/ebuf consumption.
  }
}

extern "C" void kernel_launch(void* const* d_in, const int* in_sizes, int n_in,
                              void* d_out, int out_size, void* d_ws,
                              size_t ws_size, hipStream_t stream) {
  (void)in_sizes; (void)n_in; (void)out_size; (void)ws_size;
  const int*   x   = (const int*)  d_in[0];
  const float* h0  = (const float*)d_in[1];
  const float* emb = (const float*)d_in[2];
  const float* Wxh = (const float*)d_in[3];
  const float* Whh = (const float*)d_in[4];
  const float* bh  = (const float*)d_in[5];
  const float* Why = (const float*)d_in[6];
  const float* by  = (const float*)d_in[7];

  float* logits = (float*)d_out;
  float* finalh = logits + (size_t)B_ * L_ * V_;

  // ~2.81 MB of bf16 tables in workspace (L2-resident thereafter).
  bf16_t* whh_z  = (bf16_t*)d_ws;
  bf16_t* wxh_z  = whh_z + (size_t)H_ * H_;
  bf16_t* why_z  = wxh_z + (size_t)E_ * H_;
  bf16_t* emb_bf = why_z + (size_t)H_ * V_;

  swizzle_weights<<<512, 256, 0, stream>>>(Whh, whh_z, H_, H_);
  swizzle_weights<<<128, 256, 0, stream>>>(Wxh, wxh_z, E_, H_);
  swizzle_weights<<<128, 256, 0, stream>>>(Why, why_z, H_, V_);
  convert_bf16<<<64, 256, 0, stream>>>(emb, emb_bf, V_ * E_);

  size_t smem = (size_t)B_ * HS * sizeof(bf16_t) +
                (size_t)B_ * ES * sizeof(bf16_t);        // ~146 KB of LDS
  (void)hipFuncSetAttribute((const void*)rnn_fused,
                            hipFuncAttributeMaxDynamicSharedMemorySize,
                            (int)smem);
  rnn_fused<<<1, 1024, smem, stream>>>(x, h0, emb_bf, wxh_z, whh_z, bh,
                                       why_z, by, logits, finalh);
}
